// SelfAttention_47622597378305
// MI455X (gfx1250) — matrix operand order repaired and checked by reading, not compiled, not run on hardware
//
#include <hip/hip_runtime.h>
#include <hip/hip_bf16.h>

typedef __attribute__((ext_vector_type(16))) _Float16 v16h;
typedef __attribute__((ext_vector_type(8)))  _Float16 v8h;
typedef __attribute__((ext_vector_type(8)))  float    v8f;

#define S_LEN 2048
#define NH    16
#define DH    64
#define ROWSTRIDE (3 * NH * DH)   // floats between consecutive sequence positions
#define NEG_BIG -10000.0f
#define EPS_SP  1e-6f

__device__ __forceinline__ v8f wmma16(v16h a, v16h b, v8f c) {
  // D = A(16x32 f16) * B(32x16 f16) + C(16x16 f32)
  return __builtin_amdgcn_wmma_f32_16x16x32_f16(false, a, false, b, (short)0, c,
                                                false, false);
}

__device__ __forceinline__ float rmax16(float v) {
  v = fmaxf(v, __shfl_xor(v, 1, 32));
  v = fmaxf(v, __shfl_xor(v, 2, 32));
  v = fmaxf(v, __shfl_xor(v, 4, 32));
  v = fmaxf(v, __shfl_xor(v, 8, 32));
  return v;
}
__device__ __forceinline__ float rsum16(float v) {
  v += __shfl_xor(v, 1, 32);
  v += __shfl_xor(v, 2, 32);
  v += __shfl_xor(v, 4, 32);
  v += __shfl_xor(v, 8, 32);
  return v;
}

// A-matrix (16x32 f16) per-lane fill: halves 0..7 <- row[d0..d0+7],
// halves 8..15 <- row[d0+8..d0+15]  (d0 already includes 16*lhi)
__device__ __forceinline__ v16h load_qa(const float* row, int d0, float sc) {
  float4 f0 = *(const float4*)(row + d0);
  float4 f1 = *(const float4*)(row + d0 + 4);
  float4 f2 = *(const float4*)(row + d0 + 8);
  float4 f3 = *(const float4*)(row + d0 + 12);
  v16h r;
  r[0]  = (_Float16)(f0.x * sc); r[1]  = (_Float16)(f0.y * sc);
  r[2]  = (_Float16)(f0.z * sc); r[3]  = (_Float16)(f0.w * sc);
  r[4]  = (_Float16)(f1.x * sc); r[5]  = (_Float16)(f1.y * sc);
  r[6]  = (_Float16)(f1.z * sc); r[7]  = (_Float16)(f1.w * sc);
  r[8]  = (_Float16)(f2.x * sc); r[9]  = (_Float16)(f2.y * sc);
  r[10] = (_Float16)(f2.z * sc); r[11] = (_Float16)(f2.w * sc);
  r[12] = (_Float16)(f3.x * sc); r[13] = (_Float16)(f3.y * sc);
  r[14] = (_Float16)(f3.z * sc); r[15] = (_Float16)(f3.w * sc);
  return r;
}

// B-matrix fill from 16 contiguous floats (K = h + 16*lhi already folded into p)
__device__ __forceinline__ v16h load_kb(const float* p) {
  float4 f0 = ((const float4*)p)[0];
  float4 f1 = ((const float4*)p)[1];
  float4 f2 = ((const float4*)p)[2];
  float4 f3 = ((const float4*)p)[3];
  v16h r;
  r[0]  = (_Float16)f0.x; r[1]  = (_Float16)f0.y;
  r[2]  = (_Float16)f0.z; r[3]  = (_Float16)f0.w;
  r[4]  = (_Float16)f1.x; r[5]  = (_Float16)f1.y;
  r[6]  = (_Float16)f1.z; r[7]  = (_Float16)f1.w;
  r[8]  = (_Float16)f2.x; r[9]  = (_Float16)f2.y;
  r[10] = (_Float16)f2.z; r[11] = (_Float16)f2.w;
  r[12] = (_Float16)f3.x; r[13] = (_Float16)f3.y;
  r[14] = (_Float16)f3.z; r[15] = (_Float16)f3.w;
  return r;
}

// V B-matrix fill: 16 keys (rows) strided by ROWSTRIDE floats, fixed column d
__device__ __forceinline__ v16h load_vb(const float* p) {
  v16h r;
#pragma unroll
  for (int i = 0; i < 16; ++i)
    r[i] = (_Float16)p[(size_t)i * ROWSTRIDE];
  return r;
}

__global__ __launch_bounds__(128) void softpick_attn_kernel(
    const float* __restrict__ qkv, float* __restrict__ out) {
  __shared__ __attribute__((aligned(16))) _Float16 pbuf_all[4][16 * 32];

  const int lane = threadIdx.x & 31;
  const int wv   = threadIdx.x >> 5;
  const int lhi  = lane >> 4;     // 0: lanes 0-15, 1: lanes 16-31
  const int ln   = lane & 15;
  _Float16* pb = &pbuf_all[wv][0];

  // 2048 q-tiles of 32 rows: tile = bh*64 + q32
  const int tile  = blockIdx.x * 4 + wv;
  const int q32   = tile & 63;
  const int bh    = tile >> 6;
  const int b     = bh >> 4;
  const int hh    = bh & 15;
  const int qbase = q32 * 32;

  const float sc = 0.125f;  // 1/sqrt(64), folded into Q

  // ---- Two Q sub-tiles (rows qbase..+15 and qbase+16..+31) in A-layout ----
  const float* qrowA =
      qkv + ((size_t)(b * S_LEN + (qbase + ln)) * 3 + 0) * (NH * DH) + hh * DH;
  const float* qrowB = qrowA + (size_t)16 * ROWSTRIDE;
  v16h aqA0 = load_qa(qrowA, lhi * 16, sc);
  v16h aqA1 = load_qa(qrowA, 32 + lhi * 16, sc);
  v16h aqB0 = load_qa(qrowB, lhi * 16, sc);
  v16h aqB1 = load_qa(qrowB, 32 + lhi * 16, sc);

  v8f oA0 = {}, oA1 = {}, oA2 = {}, oA3 = {};
  v8f oB0 = {}, oB1 = {}, oB2 = {}, oB3 = {};
  float mA[8], dA[8], mB[8], dB[8];
#pragma unroll
  for (int r = 0; r < 8; ++r) {
    mA[r] = -1e30f; dA[r] = 0.0f;
    mB[r] = -1e30f; dB[r] = 0.0f;
  }

  const int kend = qbase + 32;  // causal: keys 0..qbase+31, tiles of 32

  const float* kbp =
      qkv + ((size_t)(b * S_LEN) * 3 + 1) * (NH * DH) + hh * DH + lhi * 16;
  const float* vbp =
      qkv + ((size_t)(b * S_LEN) * 3 + 2) * (NH * DH) + hh * DH + ln;

  for (int kb = 0; kb < kend; kb += 32) {
    const float* k0p = kbp + (size_t)(kb + ln) * ROWSTRIDE;
    const float* k1p = k0p + (size_t)16 * ROWSTRIDE;
    const float* vp  = vbp + (size_t)(kb + lhi * 16) * ROWSTRIDE;

    // prefetch next key tile while this iteration computes (global_prefetch_b8)
    __builtin_prefetch(k0p + (size_t)32 * ROWSTRIDE, 0, 3);
    __builtin_prefetch(k1p + (size_t)32 * ROWSTRIDE, 0, 3);
    __builtin_prefetch(vp  + (size_t)32 * ROWSTRIDE, 0, 3);

    // ---- K B-regs shared by both q sub-tiles ----
    v16h bk00 = load_kb(k0p);
    v16h bk01 = load_kb(k0p + 32);
    v16h bk10 = load_kb(k1p);
    v16h bk11 = load_kb(k1p + 32);

    // ---- scores: 4 tiles (2 q sub-tiles x 2 key sub-tiles) ----
    v8f sA0 = {}, sA1 = {}, sB0 = {}, sB1 = {};
    sA0 = wmma16(aqA0, bk00, sA0); sA0 = wmma16(aqA1, bk01, sA0);
    sA1 = wmma16(aqA0, bk10, sA1); sA1 = wmma16(aqA1, bk11, sA1);
    sB0 = wmma16(aqB0, bk00, sB0); sB0 = wmma16(aqB1, bk01, sB0);
    sB1 = wmma16(aqB0, bk10, sB1); sB1 = wmma16(aqB1, bk11, sB1);

    // ---- V B-regs shared by both q sub-tiles ----
    v16h vb0 = load_vb(vp +  0);
    v16h vb1 = load_vb(vp + 16);
    v16h vb2 = load_vb(vp + 32);
    v16h vb3 = load_vb(vp + 48);

    const int k0i = kb + ln;
    const int k1i = kb + 16 + ln;

    // softpick online update + P@V for one 16-row q sub-tile
    auto step = [&](v8f s0, v8f s1, const int qib,
                    float (&m)[8], float (&den)[8],
                    v8f& o0, v8f& o1, v8f& o2, v8f& o3) {
      float alpha[8];
#pragma unroll
      for (int r = 0; r < 8; ++r) {
        const int qi = qib + r;
        float x0 = (k0i <= qi) ? s0[r] : NEG_BIG;  // finite mask, like reference
        float x1 = (k1i <= qi) ? s1[r] : NEG_BIG;
        float mx = rmax16(fmaxf(x0, x1));
        float mn = fmaxf(m[r], mx);
        float al = __expf(m[r] - mn);
        float em = __expf(-mn);
        float e0 = __expf(x0 - mn) - em;
        float e1 = __expf(x1 - mn) - em;
        float ab = rsum16(fabsf(e0) + fabsf(e1));
        den[r] = den[r] * al + ab;
        m[r] = mn;
        alpha[r] = al;
        pb[(r + 8 * lhi) * 32 + ln]      = (_Float16)fmaxf(e0, 0.0f);
        pb[(r + 8 * lhi) * 32 + 16 + ln] = (_Float16)fmaxf(e1, 0.0f);
      }
      // drain this wave's cross-lane LDS traffic before re-reading
      asm volatile("s_wait_dscnt 0x0" ::: "memory");
      v8h plo = *(const v8h*)(pb + ln * 32 + lhi * 16);
      v8h phi = *(const v8h*)(pb + ln * 32 + lhi * 16 + 8);
      v16h pa;
#pragma unroll
      for (int i = 0; i < 8; ++i) { pa[i] = plo[i]; pa[i + 8] = phi[i]; }
#pragma unroll
      for (int r = 0; r < 8; ++r) {
        o0[r] *= alpha[r]; o1[r] *= alpha[r];
        o2[r] *= alpha[r]; o3[r] *= alpha[r];
      }
      o0 = wmma16(pa, vb0, o0);
      o1 = wmma16(pa, vb1, o1);
      o2 = wmma16(pa, vb2, o2);
      o3 = wmma16(pa, vb3, o3);
    };

    step(sA0, sA1, qbase + lhi * 8,      mA, dA, oA0, oA1, oA2, oA3);
    step(sB0, sB1, qbase + 16 + lhi * 8, mB, dB, oB0, oB1, oB2, oB3);
  }

  // ---- epilogue: tail masked keys contribute ~exp(-m) each to denominator ----
  const float tail = (float)(S_LEN - kend);

  auto finish = [&](float (&m)[8], float (&den)[8], int row0,
                    v8f& o0, v8f& o1, v8f& o2, v8f& o3) {
#pragma unroll
    for (int r = 0; r < 8; ++r) {
      float d = den[r] + tail * __expf(-m[r]) + EPS_SP;
      float inv = 1.0f / d;
      float* op = out +
          ((size_t)(b * S_LEN + row0 + r + 8 * lhi) * NH + hh) * DH + ln;
      op[0]  = o0[r] * inv;
      op[16] = o1[r] * inv;
      op[32] = o2[r] * inv;
      op[48] = o3[r] * inv;
    }
  };

  finish(mA, dA, qbase,      oA0, oA1, oA2, oA3);
  finish(mB, dB, qbase + 16, oB0, oB1, oB2, oB3);
}

extern "C" void kernel_launch(void* const* d_in, const int* in_sizes, int n_in,
                              void* d_out, int out_size, void* d_ws, size_t ws_size,
                              hipStream_t stream) {
  (void)in_sizes; (void)n_in; (void)d_ws; (void)ws_size; (void)out_size;
  const float* qkv = (const float*)d_in[0];
  float* out = (float*)d_out;
  // 2*16*64 = 2048 q-tiles (32 rows each), 4 waves (128 threads) per block
  softpick_attn_kernel<<<dim3(512), dim3(128), 0, stream>>>(qkv, out);
}
